// SparseLinear_8856222564675
// MI455X (gfx1250) — compile-verified, matrix operand order checked by reading, not run on hardware
//
#include <hip/hip_runtime.h>
#include <hip/hip_bf16.h>
#include <stdint.h>

// ---------------------------------------------------------------------------
// SparseLinear (50% random density) on MI455X:
//   densify CSR -> bf16 W (32MB, L2-resident), then dense WMMA GEMM
//   Y[16384,4096] = Xbf16 @ Wbf16^T + bias, fp32 accumulate.
//   GEMM: block 256x128, BK=64, async global->LDS double buffering,
//         8 waves (4x2), wave tile 64x64 = 16x v_wmma_f32_16x16x32_bf16 / ksub.
// ---------------------------------------------------------------------------

typedef __attribute__((ext_vector_type(16))) __bf16 v16bf;
typedef __attribute__((ext_vector_type(8)))  float  v8f;
typedef __attribute__((ext_vector_type(4)))  __bf16 v4bf;
typedef __attribute__((ext_vector_type(4)))  int    v4i;

#define IN_F   4096
#define OUT_F  4096
#define NTOK   16384

// GEMM tiling
#define BM  256
#define BN  128
#define BK  64
#define LDT 72                              // padded LDS row stride (bf16 elems)
#define A_TILE_BYTES (BM * LDT * 2)         // 36864
#define B_TILE_BYTES (BN * LDT * 2)         // 18432
#define SMEM_BYTES   (2 * (A_TILE_BYTES + B_TILE_BYTES))  // 110592 (<320KB/WGP)

union FragBF16 { v16bf v; uint4 u[2]; };

// --- CDNA5 async global->LDS copy (ASYNCcnt) with safe fallback ------------
#if defined(__has_builtin)
#if __has_builtin(__builtin_amdgcn_global_load_async_to_lds_b128)
#define HAVE_ASYNC_B128 1
#endif
#if __has_builtin(__builtin_amdgcn_s_wait_asynccnt)
#define HAVE_WAIT_ASYNC 1
#endif
#endif

#if defined(__AMDGCN__)
typedef __attribute__((address_space(1))) v4i* gv4i_p;   // global int4*
typedef __attribute__((address_space(3))) v4i* sv4i_p;   // LDS int4*
#endif

__device__ __forceinline__ void async_copy16(const void* g, void* l) {
#if defined(HAVE_ASYNC_B128) && defined(__AMDGCN__)
    __builtin_amdgcn_global_load_async_to_lds_b128(
        (gv4i_p)g, (sv4i_p)l, /*offset=*/0, /*cpol=*/0);
#else
    *(uint4*)l = *(const uint4*)g;
#endif
}

__device__ __forceinline__ void wait_async_all() {
#if defined(HAVE_ASYNC_B128) && defined(__AMDGCN__)
#if defined(HAVE_WAIT_ASYNC)
    __builtin_amdgcn_s_wait_asynccnt(0);
#else
    asm volatile("s_wait_asynccnt 0x0" ::: "memory");
#endif
#endif
}

// --- 1) zero the dense-W workspace region (16B stores) ---------------------
__global__ __launch_bounds__(256) void zero_w_kernel(uint4* __restrict__ wb4) {
    size_t i = (size_t)blockIdx.x * blockDim.x + threadIdx.x;
    uint4 z; z.x = 0u; z.y = 0u; z.z = 0u; z.w = 0u;
    wb4[i] = z;
}

// --- 2) X fp32 -> bf16 (streaming, NT stores) ------------------------------
__global__ __launch_bounds__(256) void cvt_x_kernel(const float4* __restrict__ x4,
                                                    v4bf* __restrict__ xb4) {
    size_t i = (size_t)blockIdx.x * blockDim.x + threadIdx.x;
    float4 f = x4[i];
    v4bf b;
    b.x = (__bf16)f.x;
    b.y = (__bf16)f.y;
    b.z = (__bf16)f.z;
    b.w = (__bf16)f.w;
    __builtin_nontemporal_store(b, &xb4[i]);
}

// --- 3) scatter CSR values into dense bf16 W (unique (row,col): race-free) -
__global__ __launch_bounds__(256) void scatter_w_kernel(const float* __restrict__ vals,
                                                        const int* __restrict__ rows,
                                                        const int* __restrict__ cols,
                                                        __bf16* __restrict__ wb,
                                                        int nnz) {
    int i = blockIdx.x * blockDim.x + threadIdx.x;
    if (i < nnz)
        wb[(size_t)rows[i] * IN_F + cols[i]] = (__bf16)vals[i];
}

// --- 4) WMMA GEMM ----------------------------------------------------------
__global__ __launch_bounds__(256) void sparse_linear_wmma_kernel(
    const __bf16* __restrict__ Xb,   // [NTOK,  IN_F] row-major bf16
    const __bf16* __restrict__ Wb,   // [OUT_F, IN_F] row-major bf16
    const float*  __restrict__ bias, // [OUT_F]
    float*        __restrict__ Y)    // [NTOK, OUT_F] row-major fp32
{
    extern __shared__ char smem[];
    char* const smA0 = smem;
    char* const smA1 = smem + A_TILE_BYTES;
    char* const smB0 = smem + 2 * A_TILE_BYTES;
    char* const smB1 = smem + 2 * A_TILE_BYTES + B_TILE_BYTES;

    const int tid    = threadIdx.x;
    const int lane   = tid & 31;
    const int wave   = tid >> 5;
    const int waveM  = wave >> 1;            // 0..3 -> M offset 0..192
    const int waveN  = wave & 1;             // 0..1 -> N offset 0/64
    const int blockM = blockIdx.y * BM;
    const int blockN = blockIdx.x * BN;
    const int wMo    = waveM * 64;           // within-block wave M base
    const int wNo    = waveN * 64;           // within-block wave N base
    const int half   = lane >> 4;            // K-halfgroup held by this lane
    const int r      = lane & 15;

    // ---- copy-phase per-thread chunk coordinates (16B granularity) ----
    // thread t covers rows (t>>3)+32*it, column chunk (t&7)*8 of the BK slab
    const int crow = tid >> 3;               // 0..31
    const int ccol = (tid & 7) * 8;          // bf16 element offset in [0,BK)
    const __bf16* const gA = Xb + (size_t)(blockM + crow) * IN_F + ccol;
    const __bf16* const gB = Wb + (size_t)(blockN + crow) * IN_F + ccol;
    const int lOff = (crow * LDT + ccol) * 2;        // byte offset in tile

    auto copy_tiles = [&](int k, char* sA, char* sB) {
#pragma unroll
        for (int it = 0; it < 8; ++it)               // A: 256 rows
            async_copy16(gA + (size_t)(32 * it) * IN_F + k,
                         sA + lOff + 32 * it * LDT * 2);
#pragma unroll
        for (int it = 0; it < 4; ++it)               // B: 128 rows
            async_copy16(gB + (size_t)(32 * it) * IN_F + k,
                         sB + lOff + 32 * it * LDT * 2);
    };

    v8f zero = {};
    v8f acc[4][4];
#pragma unroll
    for (int i = 0; i < 4; ++i)
#pragma unroll
        for (int j = 0; j < 4; ++j)
            acc[i][j] = zero;

    // fragment LDS byte offsets (row*LDT + ks + half*8 elems; +32B for chunk1)
    const int aFragOff = ((wMo + r) * LDT + half * 8) * 2;
    const int bFragOff = ((wNo + r) * LDT + half * 8) * 2;

    auto compute = [&](const char* sA, const char* sB) {
#pragma unroll
        for (int ks = 0; ks < BK; ks += 32) {
            FragBF16 a[4], b[4];
#pragma unroll
            for (int i = 0; i < 4; ++i) {
                const char* p = sA + aFragOff + (i * 16 * LDT + ks) * 2;
                a[i].u[0] = *(const uint4*)p;
                a[i].u[1] = *(const uint4*)(p + 32);
            }
#pragma unroll
            for (int j = 0; j < 4; ++j) {
                const char* p = sB + bFragOff + (j * 16 * LDT + ks) * 2;
                b[j].u[0] = *(const uint4*)p;
                b[j].u[1] = *(const uint4*)(p + 32);
            }
#pragma unroll
            for (int i = 0; i < 4; ++i)
#pragma unroll
                for (int j = 0; j < 4; ++j)
                    acc[i][j] = __builtin_amdgcn_wmma_f32_16x16x32_bf16(
                        false, a[i].v, false, b[j].v,
                        (short)0, acc[i][j], false, false);
        }
    };

    // ---- async double-buffered main loop (2 slabs per iteration) ----
    copy_tiles(0, smA0, smB0);
    wait_async_all();
    __syncthreads();

    for (int k = 0; k < IN_F; k += 2 * BK) {
        copy_tiles(k + BK, smA1, smB1);          // k+BK <= 4032 < IN_F always
        compute(smA0, smB0);
        wait_async_all();
        __syncthreads();

        if (k + 2 * BK < IN_F)
            copy_tiles(k + 2 * BK, smA0, smB0);
        compute(smA1, smB1);
        wait_async_all();
        __syncthreads();
    }

    // ---- epilogue: bias add + NT stores (don't evict W from L2) ----
#pragma unroll
    for (int j = 0; j < 4; ++j) {
        const int   col = blockN + wNo + j * 16 + r;
        const float bv  = bias[col];
#pragma unroll
        for (int i = 0; i < 4; ++i) {
#pragma unroll
            for (int e = 0; e < 8; ++e) {
                const int    row = blockM + wMo + i * 16 + half * 8 + e;
                const size_t off = (size_t)row * OUT_F + col;
                __builtin_nontemporal_store(acc[i][j][e] + bv, &Y[off]);
            }
        }
    }
}

extern "C" void kernel_launch(void* const* d_in, const int* in_sizes, int n_in,
                              void* d_out, int out_size, void* d_ws, size_t ws_size,
                              hipStream_t stream) {
    const float* X      = (const float*)d_in[0];
    const float* values = (const float*)d_in[1];
    const int*   rows   = (const int*)d_in[2];
    const int*   cols   = (const int*)d_in[3];
    const float* bias   = (const float*)d_in[4];
    float*       Y      = (float*)d_out;
    const int    nnz    = in_sizes[1];

    // Workspace layout: [ W bf16 : 32MB ][ X bf16 : 128MB ]
    __bf16* Wb = (__bf16*)d_ws;
    __bf16* Xb = (__bf16*)((char*)d_ws + (size_t)OUT_F * IN_F * sizeof(__bf16));

    {   // 1) zero dense W
        const int threads = 256;
        const int blocks  = (int)(((size_t)OUT_F * IN_F * 2 / 16) / threads); // 8192
        zero_w_kernel<<<blocks, threads, 0, stream>>>((uint4*)Wb);
    }
    {   // 2) convert X
        const int threads = 256;
        const int blocks  = (int)(((size_t)NTOK * IN_F / 4) / threads); // 65536
        cvt_x_kernel<<<blocks, threads, 0, stream>>>((const float4*)X, (v4bf*)Xb);
    }
    {   // 3) scatter CSR -> dense bf16 W
        const int threads = 256;
        const int blocks  = (nnz + threads - 1) / threads;
        scatter_w_kernel<<<blocks, threads, 0, stream>>>(values, rows, cols, Wb, nnz);
    }
    {   // 4) GEMM: grid.x over N (A-tile L2 reuse; W resident in L2)
        dim3 grid(OUT_F / BN, NTOK / BM); // (32, 64)
        sparse_linear_wmma_kernel<<<grid, 256, SMEM_BYTES, stream>>>(Xb, Wb, bias, Y);
    }
}
